// HistoryFilterWeightedGCN_3917010174746
// MI455X (gfx1250) — matrix operand order repaired and checked by reading, not compiled, not run on hardware
//
#include <hip/hip_runtime.h>

typedef _Float16 h16;
typedef __attribute__((ext_vector_type(16))) _Float16 v16h;
typedef __attribute__((ext_vector_type(8)))  _Float16 v8h;
typedef __attribute__((ext_vector_type(8)))  float    v8f;

#define N_S   50000
#define N_A   10000
#define HID   64
#define E_A2S 400000
#define E_S2S 800000

// ---------------------------------------------------------------- device utils

// branch-free activations on the hardware transcendental unit
__device__ __forceinline__ float tanh_fast(float v) {
  // tanh(v) = 1 - 2/(1+exp(2v));  exp2(inf)->inf => rcp->0 => +1 ; exp2(-inf)->0 => -1
  float e = __builtin_amdgcn_exp2f(v * 2.885390081777927f);   // 2*log2(e)
  return 1.0f - 2.0f * __builtin_amdgcn_rcpf(1.0f + e);
}
__device__ __forceinline__ float sigmoid_fast(float v) {
  return __builtin_amdgcn_rcpf(1.0f + __builtin_amdgcn_exp2f(-1.4426950408889634f * v));
}

__device__ __forceinline__ v8f wmma32(v16h a, v16h b, v8f c) {
  // D = A(16x32 f16) * B(32x16 f16) + C(16x16 f32)
  return __builtin_amdgcn_wmma_f32_16x16x32_f16(false, a, false, b, (short)0, c,
                                                false, false);
}

// A fragment (16x32 f16) from row-major LDS [16][kpad].
// ISA layout: lanes 0-15 hold K in [0,8)+[16,24); lanes 16-31 hold K in [8,16)+[24,32).
__device__ __forceinline__ v16h load_a(const h16* in, int kpad) {
  const int lane = threadIdx.x & 31;
  const int m = lane & 15, hs = lane >> 4;
  const h16* row = in + m * kpad + 8 * hs;
  v8h lo = *(const v8h*)(row);        // K = 8*hs .. 8*hs+7
  v8h hi = *(const v8h*)(row + 16);   // K = 16+8*hs .. 16+8*hs+7
  v16h a;
#pragma unroll
  for (int i = 0; i < 8; ++i) { a[i] = lo[i]; a[8 + i] = hi[i]; }
  return a;
}

// B fragment (32x16 f16) from LDS weights packed column-major [64][kpad].
// ISA layout: lanes 0-15 hold K in [0,8)+[16,24); lanes 16-31 hold K in [8,16)+[24,32).
__device__ __forceinline__ v16h load_b(const h16* W, int kpad, int nt, int k0) {
  const int lane = threadIdx.x & 31;
  const int n = lane & 15, hs = lane >> 4;
  const h16* p = W + (nt * 16 + n) * kpad + k0 + 8 * hs;
  v8h lo = *(const v8h*)(p);
  v8h hi = *(const v8h*)(p + 16);
  v16h b;
#pragma unroll
  for (int i = 0; i < 8; ++i) { b[i] = lo[i]; b[8 + i] = hi[i]; }
  return b;
}

template<int KPAD>
__device__ __forceinline__ v8f layer_acc(const h16* in, const h16* W, int nt) {
  v8f c = {};
#pragma unroll
  for (int k0 = 0; k0 < KPAD; k0 += 32)
    c = wmma32(load_a(in + k0, KPAD), load_b(W, KPAD, nt, k0), c);
  return c;
}

// full 16x64 layer with bias + tanh, output restaged row-major to LDS [16][64] f16
template<int KPAD>
__device__ __forceinline__ void layer_tanh(const h16* in, const h16* W,
                                           const float* bias, h16* out) {
  const int lane = threadIdx.x & 31;
  const int n = lane & 15, hs = lane >> 4;
#pragma unroll
  for (int nt = 0; nt < 4; ++nt) {
    v8f c = layer_acc<KPAD>(in, W, nt);
    const float bb = bias[nt * 16 + n];
#pragma unroll
    for (int r = 0; r < 8; ++r)
      out[(r + 8 * hs) * 64 + nt * 16 + n] = (h16)tanh_fast(c[r] + bb);
  }
}

// first layer when the A fragment lives entirely in registers (KPAD == 32)
__device__ __forceinline__ void layer_regA_tanh(v16h a, const h16* W,
                                                const float* bias, h16* out) {
  const int lane = threadIdx.x & 31;
  const int n = lane & 15, hs = lane >> 4;
#pragma unroll
  for (int nt = 0; nt < 4; ++nt) {
    v8f c = {};
    c = wmma32(a, load_b(W, 32, nt, 0), c);
    const float bb = bias[nt * 16 + n];
#pragma unroll
    for (int r = 0; r < 8; ++r)
      out[(r + 8 * hs) * 64 + nt * 16 + n] = (h16)tanh_fast(c[r] + bb);
  }
}

// stage f32 weights [din][64] -> LDS f16 packed [64][kpad] (column-major, zero pad)
template<int KPAD>
__device__ __forceinline__ void stage_w(const float* W, int din, h16* dst,
                                        int tid, int nthr) {
  for (int idx = tid; idx < 64 * KPAD; idx += nthr) {
    int n = idx / KPAD, k = idx - n * KPAD;
    dst[idx] = (h16)((k < din) ? W[k * 64 + n] : 0.0f);
  }
}

// ---------------------------------------------------------------- kernels

__global__ void k_zero(float* p, long n) {
  long i = (long)blockIdx.x * blockDim.x + threadIdx.x;
  const long stride = (long)gridDim.x * blockDim.x;
  for (; i < n; i += stride) p[i] = 0.0f;
}

// enc = MLP3(u)  (tanh, tanh, linear), rows = N_A, input dim 8
__global__ __launch_bounds__(128) void k_encode_u(
    const float* __restrict__ u,
    const float* W1, const float* b1, const float* W2, const float* b2,
    const float* W3, const float* b3, h16* __restrict__ enc, int nTiles) {
  __shared__ __align__(16) h16 sW1[64 * 32];
  __shared__ __align__(16) h16 sW2[64 * 64];
  __shared__ __align__(16) h16 sW3[64 * 64];
  __shared__ float sB1[64], sB2[64], sB3[64];
  __shared__ __align__(16) h16 sM1[4][16 * 64];
  __shared__ __align__(16) h16 sM2[4][16 * 64];
  const int tid = threadIdx.x;
  stage_w<32>(W1, 8, sW1, tid, 128);
  stage_w<64>(W2, 64, sW2, tid, 128);
  stage_w<64>(W3, 64, sW3, tid, 128);
  for (int i = tid; i < 64; i += 128) { sB1[i] = b1[i]; sB2[i] = b2[i]; sB3[i] = b3[i]; }
  __syncthreads();
  const int wave = tid >> 5, lane = tid & 31;
  const int tile = blockIdx.x * 4 + wave;
  if (tile >= nTiles) return;
  const int r0 = tile * 16;
  const int m = lane & 15, hs = lane >> 4, n = m;
  v16h a = {};
  if (hs == 0) {
#pragma unroll
    for (int i = 0; i < 8; ++i) a[i] = (h16)u[(r0 + m) * 8 + i];  // K=0..7
  }
  layer_regA_tanh(a, sW1, sB1, sM1[wave]);
  layer_tanh<64>(sM1[wave], sW2, sB2, sM2[wave]);
#pragma unroll
  for (int nt = 0; nt < 4; ++nt) {
    v8f c = layer_acc<64>(sM2[wave], sW3, nt);
    const float bb = sB3[nt * 16 + n];
#pragma unroll
    for (int r = 0; r < 8; ++r)
      enc[(long)(r0 + r + 8 * hs) * 64 + nt * 16 + n] = (h16)(c[r] + bb);
  }
}

// enc = MLP3(concat(x, h))  rows = N_S, input dim 72 (pad 96)
__global__ __launch_bounds__(128) void k_encode_x(
    const float* __restrict__ x, const float* __restrict__ h,
    const float* W1, const float* b1, const float* W2, const float* b2,
    const float* W3, const float* b3, h16* __restrict__ enc, int nTiles) {
  __shared__ __align__(16) h16 sW1[64 * 96];
  __shared__ __align__(16) h16 sW2[64 * 64];
  __shared__ __align__(16) h16 sW3[64 * 64];
  __shared__ float sB1[64], sB2[64], sB3[64];
  __shared__ __align__(16) h16 sIn[4][16 * 96];
  __shared__ __align__(16) h16 sM1[4][16 * 64];
  __shared__ __align__(16) h16 sM2[4][16 * 64];
  const int tid = threadIdx.x;
  stage_w<96>(W1, 72, sW1, tid, 128);
  stage_w<64>(W2, 64, sW2, tid, 128);
  stage_w<64>(W3, 64, sW3, tid, 128);
  for (int i = tid; i < 64; i += 128) { sB1[i] = b1[i]; sB2[i] = b2[i]; sB3[i] = b3[i]; }
  __syncthreads();
  const int wave = tid >> 5, lane = tid & 31;
  const int tile = blockIdx.x * 4 + wave;
  if (tile >= nTiles) return;
  const int r0 = tile * 16;
  h16* in = sIn[wave];
  for (int i = lane; i < 16 * 96; i += 32) {
    int row = i / 96, k = i - row * 96;
    long s = r0 + row;
    float v = 0.0f;
    if (k < 8)       v = x[s * 8 + k];
    else if (k < 72) v = h[s * 64 + (k - 8)];
    in[i] = (h16)v;
  }
  layer_tanh<96>(in, sW1, sB1, sM1[wave]);
  layer_tanh<64>(sM1[wave], sW2, sB2, sM2[wave]);
  const int n = lane & 15, hs = lane >> 4;
#pragma unroll
  for (int nt = 0; nt < 4; ++nt) {
    v8f c = layer_acc<64>(sM2[wave], sW3, nt);
    const float bb = sB3[nt * 16 + n];
#pragma unroll
    for (int r = 0; r < 8; ++r)
      enc[(long)(r0 + r + 8 * hs) * 64 + nt * 16 + n] = (h16)(c[r] + bb);
  }
}

// per-edge: gate = sigmoid(MLP3(pos_src,pos_dst,dis)); atomic scatter gate*enc[src] to sum[dst]
__global__ __launch_bounds__(128) void k_edge(
    const float* __restrict__ posSrc, const float* __restrict__ posDst,
    const float* __restrict__ dis, const int* __restrict__ srcI,
    const int* __restrict__ dstI, const h16* __restrict__ enc,
    const float* W1, const float* b1, const float* W2, const float* b2,
    const float* W3, const float* b3,
    float* __restrict__ outSum, float* __restrict__ cnt, int nTiles) {
  __shared__ __align__(16) h16 sW1[64 * 32];
  __shared__ __align__(16) h16 sW2[64 * 64];
  __shared__ __align__(16) h16 sW3[64 * 64];
  __shared__ float sB1[64], sB2[64], sB3[64];
  __shared__ __align__(16) h16 sM1[4][16 * 64];
  __shared__ __align__(16) h16 sM2[4][16 * 64];
  __shared__ __align__(16) h16 sEnc[4][16 * 64];
  const int tid = threadIdx.x;
  stage_w<32>(W1, 5, sW1, tid, 128);
  stage_w<64>(W2, 64, sW2, tid, 128);
  stage_w<64>(W3, 64, sW3, tid, 128);
  for (int i = tid; i < 64; i += 128) { sB1[i] = b1[i]; sB2[i] = b2[i]; sB3[i] = b3[i]; }
  __syncthreads();
  const int wave = tid >> 5, lane = tid & 31;
  const int tile = blockIdx.x * 4 + wave;
  if (tile >= nTiles) return;
  const int e0 = tile * 16;
  const int m = lane & 15, hs = lane >> 4, n = m;

  // stage the 16 source encodings [16][64] f16 (b128 chunks)
  h16* myEnc = sEnc[wave];
  for (int i = lane; i < 128; i += 32) {
    int row = i >> 3, ch = i & 7;
    long s = srcI[e0 + row];
    *(v8h*)(myEnc + row * 64 + ch * 8) = *(const v8h*)(enc + s * 64 + ch * 8);
  }

  // gate-MLP input (dim 5, zero-padded to K=32) built directly as an A fragment:
  // only lanes 0-15 / K<8 are non-zero.
  v16h a = {};
  if (hs == 0) {
    int e = e0 + m;
    long s = srcI[e], d = dstI[e];
    a[0] = (h16)posSrc[2 * s];
    a[1] = (h16)posSrc[2 * s + 1];
    a[2] = (h16)posDst[2 * d];
    a[3] = (h16)posDst[2 * d + 1];
    a[4] = (h16)dis[e];
  }
  layer_regA_tanh(a, sW1, sB1, sM1[wave]);
  layer_tanh<64>(sM1[wave], sW2, sB2, sM2[wave]);

  int dI[8];
#pragma unroll
  for (int r = 0; r < 8; ++r) dI[r] = dstI[e0 + r + 8 * hs];
  if (cnt != nullptr && hs == 0) atomicAdd(&cnt[dI[0]], 1.0f);  // each of lanes 0-15 adds 1 for edge e0

#pragma unroll
  for (int nt = 0; nt < 4; ++nt) {
    v8f c = layer_acc<64>(sM2[wave], sW3, nt);
    const float bb = sB3[nt * 16 + n];
#pragma unroll
    for (int r = 0; r < 8; ++r) {
      float g = sigmoid_fast(c[r] + bb);
      float ev = (float)myEnc[(r + 8 * hs) * 64 + nt * 16 + n];
      atomicAdd(outSum + (long)dI[r] * 64 + nt * 16 + n, g * ev);
    }
  }
}

// node update: out = MLP3(concat(pos_s, h, sum_u, sum_x/max(cnt,1), x)) dim 202 (pad 224)
__global__ __launch_bounds__(64) void k_node(
    const float* __restrict__ posS, const float* __restrict__ h,
    const float* __restrict__ x, const float* __restrict__ sumU,
    const float* __restrict__ sumX, const float* __restrict__ cnt,
    const float* W1, const float* b1, const float* W2, const float* b2,
    const float* W3, const float* b3, float* __restrict__ out, int nTiles) {
  __shared__ __align__(16) h16 sW1[64 * 224];
  __shared__ __align__(16) h16 sW2[64 * 64];
  __shared__ __align__(16) h16 sW3[64 * 64];
  __shared__ float sB1[64], sB2[64], sB3[64];
  __shared__ __align__(16) h16 sIn[2][16 * 224];  // reused as mid2 after layer 1
  __shared__ __align__(16) h16 sM1[2][16 * 64];
  const int tid = threadIdx.x;
  stage_w<224>(W1, 202, sW1, tid, 64);
  stage_w<64>(W2, 64, sW2, tid, 64);
  stage_w<64>(W3, 64, sW3, tid, 64);
  for (int i = tid; i < 64; i += 64) { sB1[i] = b1[i]; sB2[i] = b2[i]; sB3[i] = b3[i]; }
  __syncthreads();
  const int wave = tid >> 5, lane = tid & 31;
  const int tile = blockIdx.x * 2 + wave;
  if (tile >= nTiles) return;
  const int r0 = tile * 16;
  h16* in = sIn[wave];
  for (int i = lane; i < 16 * 224; i += 32) {
    int row = i / 224, k = i - row * 224;
    long s = r0 + row;
    float v = 0.0f;
    if (k < 2)        v = posS[s * 2 + k];
    else if (k < 66)  v = h[s * 64 + (k - 2)];
    else if (k < 130) v = sumU[s * 64 + (k - 66)];
    else if (k < 194) v = sumX[s * 64 + (k - 130)] * __builtin_amdgcn_rcpf(fmaxf(cnt[s], 1.0f));
    else if (k < 202) v = x[s * 8 + (k - 194)];
    in[i] = (h16)v;
  }
  layer_tanh<224>(in, sW1, sB1, sM1[wave]);
  h16* mid2 = in;  // input no longer needed; reuse first 16*64 halves
  layer_tanh<64>(sM1[wave], sW2, sB2, mid2);
  const int n = lane & 15, hs = lane >> 4;
#pragma unroll
  for (int nt = 0; nt < 4; ++nt) {
    v8f c = layer_acc<64>(mid2, sW3, nt);
    const float bb = sB3[nt * 16 + n];
#pragma unroll
    for (int r = 0; r < 8; ++r)
      out[(long)(r0 + r + 8 * hs) * 64 + nt * 16 + n] = c[r] + bb;
  }
}

// ---------------------------------------------------------------- host

struct MlpP { const float *W1, *b1, *W2, *b2, *W3, *b3; };
static MlpP mk6(void* const* d_in, int base) {
  MlpP m;
  m.W1 = (const float*)d_in[base + 0]; m.b1 = (const float*)d_in[base + 1];
  m.W2 = (const float*)d_in[base + 2]; m.b2 = (const float*)d_in[base + 3];
  m.W3 = (const float*)d_in[base + 4]; m.b3 = (const float*)d_in[base + 5];
  return m;
}
static MlpP mkBlob(const float* p, int din) {
  MlpP m;
  m.W1 = p;            m.b1 = m.W1 + din * 64;
  m.W2 = m.b1 + 64;    m.b2 = m.W2 + 64 * 64;
  m.W3 = m.b2 + 64;    m.b3 = m.W3 + 64 * 64;
  return m;
}

extern "C" void kernel_launch(void* const* d_in, const int* in_sizes, int n_in,
                              void* d_out, int out_size, void* d_ws, size_t ws_size,
                              hipStream_t stream) {
  const float* h      = (const float*)d_in[0];
  const float* x      = (const float*)d_in[1];
  const float* u      = (const float*)d_in[2];
  const float* pos_s  = (const float*)d_in[3];
  const float* pos_a  = (const float*)d_in[4];
  const float* dis_a2s= (const float*)d_in[5];
  const float* dis_s2s= (const float*)d_in[6];
  const int* a2s_src  = (const int*)d_in[7];
  const int* a2s_dst  = (const int*)d_in[8];
  const int* s2s_src  = (const int*)d_in[9];
  const int* s2s_dst  = (const int*)d_in[10];

  MlpP u2h_dis, u2h_u, x2h_dis, x2h_x, hupd;
  if (n_in >= 41) {  // each MLP param tuple exposed as 6 separate arrays
    u2h_dis = mk6(d_in, 11);
    u2h_u   = mk6(d_in, 17);
    x2h_dis = mk6(d_in, 23);
    x2h_x   = mk6(d_in, 29);
    hupd    = mk6(d_in, 35);
  } else {           // each MLP param tuple flattened into one blob
    u2h_dis = mkBlob((const float*)d_in[11], 5);
    u2h_u   = mkBlob((const float*)d_in[12], 8);
    x2h_dis = mkBlob((const float*)d_in[13], 5);
    x2h_x   = mkBlob((const float*)d_in[14], 72);
    hupd    = mkBlob((const float*)d_in[15], 202);
  }

  // workspace carve (bytes)
  char* ws = (char*)d_ws;
  float* sum_u = (float*)(ws);                               // N_S*64 f32
  float* sum_x = (float*)(ws + 12800000);                    // N_S*64 f32
  float* cntv  = (float*)(ws + 25600000);                    // N_S f32
  h16*   enc_u = (h16*)  (ws + 25800192);                    // N_A*64 f16
  h16*   enc_x = (h16*)  (ws + 27080192);                    // N_S*64 f16

  // zero the accumulators (sum_u | sum_x | cnt are contiguous)
  k_zero<<<1024, 256, 0, stream>>>(sum_u, (long)N_S * 64 * 2 + N_S);

  k_encode_u<<<(N_A / 16 + 3) / 4, 128, 0, stream>>>(
      u, u2h_u.W1, u2h_u.b1, u2h_u.W2, u2h_u.b2, u2h_u.W3, u2h_u.b3,
      enc_u, N_A / 16);

  k_encode_x<<<(N_S / 16 + 3) / 4, 128, 0, stream>>>(
      x, h, x2h_x.W1, x2h_x.b1, x2h_x.W2, x2h_x.b2, x2h_x.W3, x2h_x.b3,
      enc_x, N_S / 16);

  k_edge<<<(E_A2S / 16) / 4, 128, 0, stream>>>(
      pos_a, pos_s, dis_a2s, a2s_src, a2s_dst, enc_u,
      u2h_dis.W1, u2h_dis.b1, u2h_dis.W2, u2h_dis.b2, u2h_dis.W3, u2h_dis.b3,
      sum_u, nullptr, E_A2S / 16);

  k_edge<<<(E_S2S / 16) / 4, 128, 0, stream>>>(
      pos_s, pos_s, dis_s2s, s2s_src, s2s_dst, enc_x,
      x2h_dis.W1, x2h_dis.b1, x2h_dis.W2, x2h_dis.b2, x2h_dis.W3, x2h_dis.b3,
      sum_x, cntv, E_S2S / 16);

  k_node<<<(N_S / 16 + 1) / 2, 64, 0, stream>>>(
      pos_s, h, x, sum_u, sum_x, cntv,
      hupd.W1, hupd.b1, hupd.W2, hupd.b2, hupd.W3, hupd.b3,
      (float*)d_out, N_S / 16);
}
